// GaussianUpsampling_44272522887625
// MI455X (gfx1250) — compile-verified, matrix operand order checked by reading, not run on hardware
//
#include <hip/hip_runtime.h>
#include <hip/hip_bf16.h>
#include <cstddef>

typedef __attribute__((ext_vector_type(16))) _Float16 v16h;
typedef __attribute__((ext_vector_type(8)))  _Float16 v8h;
typedef __attribute__((ext_vector_type(4)))  _Float16 v4h;
typedef __attribute__((ext_vector_type(8)))  float    v8f;

#define GU_B 32
#define GU_H 256
#define GU_N 512
#define GU_T 2048
#define TM   64            // t rows per block
#define KC   64            // k (n) chunk staged in LDS
#define XS_STRIDE 72       // f16 elements per staged row (64 + pad, 16B-aligned rows)
#define XS_BUF   (GU_H * XS_STRIDE)

#define LOG_2PI 1.8378770664093454835606594728112f
#define MASK_SCORE (-1.0e15f)

// ---------------------------------------------------------------- f32 -> f16 copy of x
__global__ __launch_bounds__(256)
void gu_cvt_kernel(const float* __restrict__ x, _Float16* __restrict__ xh) {
    const size_t i = ((size_t)blockIdx.x * 256 + threadIdx.x) * 8;
    const float4 a = *(const float4*)(x + i);
    const float4 b = *(const float4*)(x + i + 4);
    v8h h = { (_Float16)a.x, (_Float16)a.y, (_Float16)a.z, (_Float16)a.w,
              (_Float16)b.x, (_Float16)b.y, (_Float16)b.z, (_Float16)b.w };
    *(v8h*)(xh + i) = h;
}

// out[b,h,t] = softmax_n( w[b,n,t] ) . x[b,h,n]   with f16 WMMA, f32 accumulate.
template<bool USE_WS>
__global__ __launch_bounds__(256)
void gu_main_kernel(const float* __restrict__ x,     // (B,H,N) encoder_outputs f32
                    const _Float16* __restrict__ xh, // (B,H,N) f16 copy (USE_WS only)
                    const float* __restrict__ dur,   // (B,N)
                    const float* __restrict__ vars,  // (B,N)
                    const int*   __restrict__ lens,  // (B)
                    float* __restrict__ out)         // (B,H,T)
{
    __shared__ _Float16 xs[(USE_WS ? 2 : 1) * XS_BUF];    // 72 KB (dbl-buf) / 36 KB
    __shared__ float cs[GU_N], qs[GU_N], ls[GU_N];        // 6 KB gaussian params
    __shared__ float pscan[256];                          // 1 KB scan scratch

    const int tid  = threadIdx.x;
    const int lane = tid & 31;
    const int wave = tid >> 5;
    const int b    = blockIdx.y;
    const int t0   = blockIdx.x * TM;

    const int wt = wave & 3;                // 4 wave rows over t
    const int wh = wave >> 2;               // 2 wave cols over h
    const int t_wave = t0 + wt * 16;
    const int h_wave = wh * 128;

    const _Float16* xh_b = USE_WS ? (xh + (size_t)b * GU_H * GU_N) : nullptr;
    const unsigned  lds_xs = (unsigned)(size_t)&xs[0];    // LDS byte offset of xs
    const int as_h0 = tid >> 3;                           // async staging: 32 h-rows / pass
    const int as_k8 = (tid & 7) * 8;                      // 8-half (16B) column chunk

    // ---------------- per-batch params: c = cumsum(dur)-0.5*dur, q=-0.5/v, l=-0.5*(log2pi+log v)
    const int len = lens[b];
    {
        const float d0 = dur[(size_t)b * GU_N + 2 * tid];
        const float d1 = dur[(size_t)b * GU_N + 2 * tid + 1];
        pscan[tid] = d0 + d1;
        __syncthreads();
        #pragma unroll
        for (int off = 1; off < 256; off <<= 1) {
            float tpl = (tid >= off) ? pscan[tid - off] : 0.0f;
            __syncthreads();
            pscan[tid] += tpl;
            __syncthreads();
        }
        const float incl = pscan[tid];           // inclusive pair-sum up to element 2*tid+1
        cs[2 * tid]     = (incl - d1) - 0.5f * d0;
        cs[2 * tid + 1] = incl - 0.5f * d1;

        #pragma unroll
        for (int it = 0; it < 2; ++it) {
            const int n = tid + it * 256;
            const float v = vars[(size_t)b * GU_N + n];
            const bool masked = (n >= len);
            qs[n] = masked ? 0.0f : (-0.5f / v);
            ls[n] = masked ? MASK_SCORE : (-0.5f * (LOG_2PI + __logf(v)));
        }
    }
    __syncthreads();

    // ---------------- kick off async DMA of chunk 0 (overlaps with the max pass below)
    if constexpr (USE_WS) {
        #pragma unroll
        for (int it = 0; it < 8; ++it) {
            const int hh = as_h0 + it * 32;
            const _Float16* g = xh_b + (size_t)hh * GU_N + as_k8;
            const unsigned  l = lds_xs + (unsigned)((hh * XS_STRIDE + as_k8) * 2);
            asm volatile("global_load_async_to_lds_b128 %0, %1, off"
                         :: "v"(l), "v"(g) : "memory");
        }
    }

    // ---------------- pass 1: per-t row max of w over all n (for stable softmax)
    const int half = lane >> 4;              // 0: lanes 0-15, 1: lanes 16-31
    const int tcol = lane & 15;
    const float tf = (float)(t_wave + tcol);
    float mx = -3.0e38f;
    {
        const int nbase = half * 256;
        #pragma unroll 8
        for (int i = 0; i < 256; ++i) {
            const int n = nbase + i;
            const float dt = tf - cs[n];
            const float w  = __builtin_fmaf(dt * dt, qs[n], ls[n]);
            mx = fmaxf(mx, w);
        }
        mx = fmaxf(mx, __shfl_xor(mx, 16, 32));   // both lane-halves get the full-row max
    }

    // ---------------- main K loop: C[h,t] += X[h,k] * P[k,t],  P = exp(w - mx)
    float ssum = 0.0f;
    v8f acc[8];
    #pragma unroll
    for (int i = 0; i < 8; ++i) acc[i] = v8f{};

    const float* xbase = x + (size_t)b * GU_H * GU_N;

    for (int ci = 0; ci < GU_N / KC; ++ci) {
        const int kc = ci * KC;
        const _Float16* xsb;

        if constexpr (USE_WS) {
            // issue next chunk into the other buffer, then drain down to it
            if (ci < GU_N / KC - 1) {
                const int kcn = kc + KC;
                const unsigned bufoff = (unsigned)(((ci + 1) & 1) * XS_BUF * 2);
                #pragma unroll
                for (int it = 0; it < 8; ++it) {
                    const int hh = as_h0 + it * 32;
                    const _Float16* g = xh_b + (size_t)hh * GU_N + kcn + as_k8;
                    const unsigned  l = lds_xs + bufoff + (unsigned)((hh * XS_STRIDE + as_k8) * 2);
                    asm volatile("global_load_async_to_lds_b128 %0, %1, off"
                                 :: "v"(l), "v"(g) : "memory");
                }
                asm volatile("s_wait_asynccnt 0x8" ::: "memory");
            } else {
                asm volatile("s_wait_asynccnt 0x0" ::: "memory");
            }
            __syncthreads();                  // everyone's chunk ci is resident
            xsb = xs + (ci & 1) * XS_BUF;
        } else {
            __syncthreads();                  // xs consumed by previous chunk
            // stage 256h x 64k of x into LDS as f16 (coalesced 16B global loads + cvt)
            const int hh0 = tid >> 4;
            const int nl  = (tid & 15) * 4;
            #pragma unroll
            for (int it = 0; it < 16; ++it) {
                const int hh = hh0 + it * 16;
                const float4 v = *(const float4*)(xbase + (size_t)hh * GU_N + kc + nl);
                v4h hv = { (_Float16)v.x, (_Float16)v.y, (_Float16)v.z, (_Float16)v.w };
                *(v4h*)(xs + hh * XS_STRIDE + nl) = hv;
            }
            __syncthreads();
            xsb = xs;
        }

        #pragma unroll
        for (int ks = 0; ks < KC; ks += 32) {
            const int k0 = kc + ks;
            // ---- B fragment (32k x 16t): softmax numerators, computed in fragment order
            v16h bf;
            {
                const int kb = k0 + half * 8;
                #pragma unroll
                for (int j = 0; j < 8; ++j) {
                    {
                        const int n = kb + j;
                        const float dt = tf - cs[n];
                        const float w  = __builtin_fmaf(dt * dt, qs[n], ls[n]);
                        const float p  = __expf(w - mx);
                        ssum += p;
                        bf[j] = (_Float16)p;
                    }
                    {
                        const int n = kb + 16 + j;
                        const float dt = tf - cs[n];
                        const float w  = __builtin_fmaf(dt * dt, qs[n], ls[n]);
                        const float p  = __expf(w - mx);
                        ssum += p;
                        bf[j + 8] = (_Float16)p;
                    }
                }
            }
            // ---- A fragments (16h x 32k) from LDS, 8 h-tiles, 8 WMMAs
            const int kl = ks + half * 8;
            const _Float16* arow = xsb + (h_wave + tcol) * XS_STRIDE + kl;
            #pragma unroll
            for (int ht = 0; ht < 8; ++ht) {
                const _Float16* ap = arow + ht * 16 * XS_STRIDE;
                const v8h lo = *(const v8h*)(ap);
                const v8h hi = *(const v8h*)(ap + 16);
                const v16h af = __builtin_shufflevector(lo, hi,
                    0, 1, 2, 3, 4, 5, 6, 7, 8, 9, 10, 11, 12, 13, 14, 15);
                acc[ht] = __builtin_amdgcn_wmma_f32_16x16x32_f16(
                    /*neg_a=*/false, af, /*neg_b=*/false, bf,
                    /*c_mod=*/(short)0, acc[ht], /*reuse_a=*/false, /*reuse_b=*/false);
            }
        }

        if constexpr (USE_WS) {
            __syncthreads();                  // all reads of this buffer done before refill
        }
    }

    // ---------------- epilogue: divide by softmax denominator, store (B,H,T)
    ssum += __shfl_xor(ssum, 16, 32);        // combine the two k-halves per t row
    const float inv_s = 1.0f / ssum;         // lane's t == lane&15 == C lane index
    const int t_abs = t_wave + tcol;
    #pragma unroll
    for (int ht = 0; ht < 8; ++ht) {
        const int hbase = h_wave + ht * 16 + half * 8;   // C vgpr r: M = r + 8*half
        #pragma unroll
        for (int r = 0; r < 8; ++r) {
            out[((size_t)b * GU_H + (hbase + r)) * GU_T + t_abs] = acc[ht][r] * inv_s;
        }
    }
}

extern "C" void kernel_launch(void* const* d_in, const int* in_sizes, int n_in,
                              void* d_out, int out_size, void* d_ws, size_t ws_size,
                              hipStream_t stream) {
    (void)in_sizes; (void)n_in; (void)out_size;
    const float* x    = (const float*)d_in[0];   // (32,256,512) f32
    const float* dur  = (const float*)d_in[1];   // (32,512) f32
    const float* vars = (const float*)d_in[2];   // (32,512) f32
    const int*   lens = (const int*)d_in[3];     // (32) i32
    float* out = (float*)d_out;                  // (32,256,2048) f32

    const size_t xh_bytes = (size_t)GU_B * GU_H * GU_N * sizeof(_Float16);
    dim3 grid(GU_T / TM, GU_B);
    dim3 block(256);

    if (ws_size >= xh_bytes) {
        _Float16* xh = (_Float16*)d_ws;
        const size_t total = (size_t)GU_B * GU_H * GU_N;
        gu_cvt_kernel<<<(unsigned)(total / (256 * 8)), 256, 0, stream>>>(x, xh);
        gu_main_kernel<true><<<grid, block, 0, stream>>>(x, xh, dur, vars, lens, out);
    } else {
        gu_main_kernel<false><<<grid, block, 0, stream>>>(x, nullptr, dur, vars, lens, out);
    }
}
